// BeamDecoder_50637664420167
// MI455X (gfx1250) — matrix operand order repaired and checked by reading, not compiled, not run on hardware
//
#include <hip/hip_runtime.h>
#include <math.h>

// Problem constants (from reference)
#define EMBED   512
#define HIDDEN  1024
#define G4      (4 * HIDDEN)   // 4096 gate rows
#define VOCAB   50257
#define DEPTH   18             // scan steps; total decode steps = DEPTH + 1

// fc_w padded so logits grid covers it exactly: 393 blocks * 8 waves * 16 rows = 50304
#define NBLK_LOGITS 393
#define VROWS_PAD   (NBLK_LOGITS * 128)   // 50304

typedef __attribute__((ext_vector_type(16))) __bf16 v16bf;
typedef __attribute__((ext_vector_type(8)))  float  v8f;

__device__ __forceinline__ unsigned short f32_to_bf16_bits(float f) {
    unsigned int u = __float_as_uint(f);
    return (unsigned short)((u + 0x7FFFu + ((u >> 16) & 1u)) >> 16);  // RNE
}

// ---------------------------------------------------------------------------
// Weight conversion f32 -> bf16 (runs once per launch; bf16 fc_w fits in L2)
// ---------------------------------------------------------------------------
__global__ void convert_bf16_kernel(const float* __restrict__ src,
                                    unsigned short* __restrict__ dst, int n) {
    int i = blockIdx.x * blockDim.x + threadIdx.x;
    int stride = gridDim.x * blockDim.x;
    for (; i < n; i += stride) dst[i] = f32_to_bf16_bits(src[i]);
}

__global__ void convert_fcw_kernel(const float* __restrict__ src,
                                   unsigned short* __restrict__ dst) {
    int i = blockIdx.x * blockDim.x + threadIdx.x;
    int stride = gridDim.x * blockDim.x;
    const int n = VROWS_PAD * HIDDEN;
    for (; i < n; i += stride) {
        int row = i >> 10;          // / HIDDEN
        int col = i & (HIDDEN - 1);
        float v = (row < VOCAB) ? src[(size_t)row * HIDDEN + col] : 0.0f;
        dst[i] = f32_to_bf16_bits(v);
    }
}

// ---------------------------------------------------------------------------
// Init: x <- bf16(features), h_bf16 <- 0, c <- 0
// ---------------------------------------------------------------------------
__global__ void init_kernel(const float* __restrict__ features,
                            unsigned short* __restrict__ xbf,
                            unsigned short* __restrict__ hbf,
                            float* __restrict__ c) {
    int t = threadIdx.x;
    if (t < EMBED) xbf[t] = f32_to_bf16_bits(features[t]);
    if (t < HIDDEN) { hbf[t] = 0; c[t] = 0.0f; }
}

// ---------------------------------------------------------------------------
// One 16-row x 32-K bf16 WMMA GEMV tile, accumulated into acc.
// A tile layout (16-bit A 16x32, ISA 7.12.2):
//   lane<16 : row=lane,    elems 0..7 = K..K+7,   elems 8..15 = K+16..K+23
//   lane>=16: row=lane-16, elems 0..7 = K+8..K+15, elems 8..15 = K+24..K+31
// B broadcast (32x16, every column = vec): lane<16 holds K..K+7, K+16..K+23,
//   lane>=16 holds K+8..K+15, K+24..K+31 (same k order as A, column = lane%16).
// ---------------------------------------------------------------------------
__device__ __forceinline__ void wmma_gemv_accum(const unsigned short* __restrict__ W,
                                                int ld, int rowBase, int kBase,
                                                const unsigned short* __restrict__ vec,
                                                unsigned lane, v8f& acc) {
    union Pack { v16bf v; uint4 q[2]; };
    Pack A, B;
    int m    = (int)(lane & 15u);
    int koffA = (lane < 16u) ? 0 : 8;
    const unsigned short* arow = W + (size_t)(rowBase + m) * ld + kBase;
    A.q[0] = *(const uint4*)(arow + koffA);
    A.q[1] = *(const uint4*)(arow + 16 + koffA);
    int koffB = (lane < 16u) ? 0 : 8;
    const unsigned short* bp = vec + kBase + koffB;
    B.q[0] = *(const uint4*)(bp);
    B.q[1] = *(const uint4*)(bp + 16);
    acc = __builtin_amdgcn_wmma_f32_16x16x32_bf16(false, A.v, false, B.v,
                                                  (short)0, acc, false, false);
}

// ---------------------------------------------------------------------------
// gates = W_ih @ x + b_ih + W_hh @ h + b_hh   (4096 rows)
// 8 waves/block * 16 rows = 128 rows/block, 32 blocks.
// ---------------------------------------------------------------------------
__global__ void gates_kernel(const unsigned short* __restrict__ wih,
                             const unsigned short* __restrict__ whh,
                             const unsigned short* __restrict__ xbf,
                             const unsigned short* __restrict__ hbf,
                             const float* __restrict__ b_ih,
                             const float* __restrict__ b_hh,
                             float* __restrict__ gates) {
    unsigned lane = threadIdx.x & 31u;
    unsigned wave = threadIdx.x >> 5;
    int rowBase = ((int)blockIdx.x * 8 + (int)wave) * 16;
    v8f acc = {};
    #pragma unroll 4
    for (int k = 0; k < EMBED; k += 32)
        wmma_gemv_accum(wih, EMBED, rowBase, k, xbf, lane, acc);
    #pragma unroll 4
    for (int k = 0; k < HIDDEN; k += 32)
        wmma_gemv_accum(whh, HIDDEN, rowBase, k, hbf, lane, acc);
    // D layout: lane 0 -> rows rowBase+0..7 ; lane 16 -> rows rowBase+8..15 (N=0)
    if (lane == 0u) {
        #pragma unroll
        for (int v = 0; v < 8; ++v) {
            int r = rowBase + v;
            gates[r] = acc[v] + b_ih[r] + b_hh[r];
        }
    } else if (lane == 16u) {
        #pragma unroll
        for (int v = 0; v < 8; ++v) {
            int r = rowBase + 8 + v;
            gates[r] = acc[v] + b_ih[r] + b_hh[r];
        }
    }
}

// ---------------------------------------------------------------------------
// LSTM cell elementwise (PyTorch gate order i,f,g,o)
// ---------------------------------------------------------------------------
__global__ void cell_kernel(const float* __restrict__ gates,
                            float* __restrict__ c,
                            unsigned short* __restrict__ hbf) {
    int j = blockIdx.x * blockDim.x + threadIdx.x;
    if (j >= HIDDEN) return;
    float ig = gates[j];
    float fg = gates[HIDDEN + j];
    float gg = gates[2 * HIDDEN + j];
    float og = gates[3 * HIDDEN + j];
    float si = 1.0f / (1.0f + expf(-ig));
    float sf = 1.0f / (1.0f + expf(-fg));
    float so = 1.0f / (1.0f + expf(-og));
    float cn = sf * c[j] + si * tanhf(gg);
    float h  = so * tanhf(cn);
    c[j]   = cn;
    hbf[j] = f32_to_bf16_bits(h);
}

// ---------------------------------------------------------------------------
// logits = fc_w @ h + fc_b ; per-block argmax over its 128 rows -> partials.
// fc_w padded/zero-filled to VROWS_PAD rows so every wave runs WMMA with
// EXEC all-ones; pad rows masked with -inf only at the argmax stage.
// ---------------------------------------------------------------------------
__global__ void logits_kernel(const unsigned short* __restrict__ fcw,
                              const unsigned short* __restrict__ hbf,
                              const float* __restrict__ fc_b,
                              float* __restrict__ partV,
                              int* __restrict__ partI) {
    __shared__ float sval[128];
    __shared__ int   sidx[128];
    unsigned lane = threadIdx.x & 31u;
    unsigned wave = threadIdx.x >> 5;
    int rowBase = ((int)blockIdx.x * 8 + (int)wave) * 16;
    v8f acc = {};
    #pragma unroll 4
    for (int k = 0; k < HIDDEN; k += 32)
        wmma_gemv_accum(fcw, HIDDEN, rowBase, k, hbf, lane, acc);
    if (lane == 0u || lane == 16u) {
        int half = (int)(lane >> 4) * 8;
        #pragma unroll
        for (int v = 0; v < 8; ++v) {
            int r = rowBase + half + v;
            float val = (r < VOCAB) ? (acc[v] + fc_b[r]) : -__builtin_inff();
            int s = (int)wave * 16 + half + v;
            sval[s] = val;
            sidx[s] = r;
        }
    }
    __syncthreads();
    for (int st = 64; st > 0; st >>= 1) {
        if ((int)threadIdx.x < st) {
            if (sval[threadIdx.x + st] > sval[threadIdx.x]) {
                sval[threadIdx.x] = sval[threadIdx.x + st];
                sidx[threadIdx.x] = sidx[threadIdx.x + st];
            }
        }
        __syncthreads();
    }
    if (threadIdx.x == 0) {
        partV[blockIdx.x] = sval[0];
        partI[blockIdx.x] = sidx[0];
    }
}

// ---------------------------------------------------------------------------
// Global argmax over partials; record token & running logit product; prep
// next step's embedding input (bf16) for the chosen token.
// ---------------------------------------------------------------------------
__global__ void reduce_kernel(const float* __restrict__ partV,
                              const int* __restrict__ partI,
                              int nb, int t,
                              const float* __restrict__ embed,
                              unsigned short* __restrict__ xbf,
                              int* __restrict__ toks,
                              float* __restrict__ probs,
                              float* __restrict__ out) {
    __shared__ float sval[512];
    __shared__ int   sidx[512];
    __shared__ int   stok;
    int tid = threadIdx.x;
    float mv = -__builtin_inff();
    int   mi = 0;
    for (int i = tid; i < nb; i += 512) {
        if (partV[i] > mv) { mv = partV[i]; mi = partI[i]; }
    }
    sval[tid] = mv;
    sidx[tid] = mi;
    __syncthreads();
    for (int st = 256; st > 0; st >>= 1) {
        if (tid < st) {
            if (sval[tid + st] > sval[tid]) {
                sval[tid] = sval[tid + st];
                sidx[tid] = sidx[tid + st];
            }
        }
        __syncthreads();
    }
    if (tid == 0) {
        float top = sval[0];
        int   idx = sidx[0];
        toks[t]  = idx;
        probs[t] = (t == 0) ? top : probs[t - 1] * top;
        out[1 + t] = (float)idx;      // tokens[t] in output
        stok = idx;
    }
    __syncthreads();
    int tok = stok;
    if (tid < EMBED)
        xbf[tid] = f32_to_bf16_bits(embed[(size_t)tok * EMBED + tid]);
}

// ---------------------------------------------------------------------------
// Pick best depth: argmax over probs[1..DEPTH]
// ---------------------------------------------------------------------------
__global__ void final_kernel(const float* __restrict__ probs, float* __restrict__ out) {
    if (threadIdx.x == 0) {
        float best = probs[1];
        int bt = 1;
        for (int t = 2; t <= DEPTH; ++t)
            if (probs[t] > best) { best = probs[t]; bt = t; }
        out[0]  = best;        // best_seq_prob
        out[20] = (float)bt;   // best_depth
    }
}

// ---------------------------------------------------------------------------
extern "C" void kernel_launch(void* const* d_in, const int* in_sizes, int n_in,
                              void* d_out, int out_size, void* d_ws, size_t ws_size,
                              hipStream_t stream) {
    (void)in_sizes; (void)n_in; (void)out_size;
    const float* features = (const float*)d_in[0];
    const float* embed    = (const float*)d_in[1];
    const float* W_ih     = (const float*)d_in[2];
    const float* W_hh     = (const float*)d_in[3];
    const float* b_ih     = (const float*)d_in[4];
    const float* b_hh     = (const float*)d_in[5];
    const float* fc_w     = (const float*)d_in[6];
    const float* fc_b     = (const float*)d_in[7];
    float* out = (float*)d_out;

    // Workspace layout (256B aligned slabs)
    char* ws = (char*)d_ws;
    size_t off = 0;
    auto alloc = [&](size_t bytes) {
        size_t o = off;
        off = (off + bytes + 255) & ~(size_t)255;
        return o;
    };
    unsigned short* wih_bf = (unsigned short*)(ws + alloc((size_t)G4 * EMBED * 2));
    unsigned short* whh_bf = (unsigned short*)(ws + alloc((size_t)G4 * HIDDEN * 2));
    unsigned short* fcw_bf = (unsigned short*)(ws + alloc((size_t)VROWS_PAD * HIDDEN * 2));
    unsigned short* xbf    = (unsigned short*)(ws + alloc((size_t)EMBED * 2));
    unsigned short* hbf    = (unsigned short*)(ws + alloc((size_t)HIDDEN * 2));
    float*          cbuf   = (float*)(ws + alloc((size_t)HIDDEN * 4));
    float*          gates  = (float*)(ws + alloc((size_t)G4 * 4));
    float*          partV  = (float*)(ws + alloc((size_t)NBLK_LOGITS * 4));
    int*            partI  = (int*)(ws + alloc((size_t)NBLK_LOGITS * 4));
    int*            toks   = (int*)(ws + alloc((size_t)(DEPTH + 1) * 4));
    float*          probs  = (float*)(ws + alloc((size_t)(DEPTH + 1) * 4));
    if (off > ws_size) return;  // deterministic no-op if workspace too small

    // One-time per-launch bf16 weight conversion (fc_w bf16 ~103MB -> L2 resident)
    convert_bf16_kernel<<<2048, 256, 0, stream>>>(W_ih, wih_bf, G4 * EMBED);
    convert_bf16_kernel<<<2048, 256, 0, stream>>>(W_hh, whh_bf, G4 * HIDDEN);
    convert_fcw_kernel<<<4096, 256, 0, stream>>>(fc_w, fcw_bf);
    init_kernel<<<1, 1024, 0, stream>>>(features, xbf, hbf, cbuf);

    // 19 sequentially-dependent decode steps
    for (int t = 0; t <= DEPTH; ++t) {
        gates_kernel<<<32, 256, 0, stream>>>(wih_bf, whh_bf, xbf, hbf, b_ih, b_hh, gates);
        cell_kernel<<<4, 256, 0, stream>>>(gates, cbuf, hbf);
        logits_kernel<<<NBLK_LOGITS, 256, 0, stream>>>(fcw_bf, hbf, fc_b, partV, partI);
        reduce_kernel<<<1, 512, 0, stream>>>(partV, partI, NBLK_LOGITS, t,
                                             embed, xbf, toks, probs, out);
    }
    final_kernel<<<1, 64, 0, stream>>>(probs, out);
}
